// GAT_53970559041999
// MI455X (gfx1250) — compile-verified, matrix operand order checked by reading, not run on hardware
//
#include <hip/hip_runtime.h>

// ---------------------------------------------------------------------------
// 2-layer GAT (heads=1) for MI455X / gfx1250, wave32.
//   layer: h = in @ W  (bf16 WMMA, f32 accum; W transposed+packed in LDS)
//          as = h.a_src, ad = h.a_dst
//          e  = leaky_relu(as[src] + ad[dst], 0.2)
//          softmax over dst segments (atomic max + atomic sum)
//          agg[dst] += alpha * h[src]  (hardware f32 atomics), out = agg + b
//   relu between layers folded into layer-2 GEMM input load (template param).
// ---------------------------------------------------------------------------

#define D 128
#define NEG_SLOPE 0.2f

typedef __attribute__((ext_vector_type(16))) __bf16 v16bf;
typedef __attribute__((ext_vector_type(8)))  __bf16 v8bf;
typedef __attribute__((ext_vector_type(8)))  float  v8f;

// order-preserving float -> uint key (monotonic), for atomicMax-based segment max
__device__ __forceinline__ unsigned f2key(float f) {
    unsigned b = __float_as_uint(f);
    return (b & 0x80000000u) ? ~b : (b | 0x80000000u);
}
__device__ __forceinline__ float key2f(unsigned k) {
    return (k & 0x80000000u) ? __uint_as_float(k ^ 0x80000000u) : __uint_as_float(~k);
}

__device__ __forceinline__ void atomicAddF32(float* p, float v) {
    unsafeAtomicAdd(p, v);   // global_atomic_add_f32
}

// ---------------------------------------------------------------------------
// GEMM: out[N x 128] = (RELU ? relu(in) : in) @ W[128 x 128]
// 256 threads = 8 waves; block covers 128 rows.
// LDS holds W^T as bf16, row stride 136 halfs (16B-aligned, bank-spread):
//   sWT[n * 136 + k] = W[k][n]
// 16-bit B fragment (ISA 7.12.2, mirrored A layout): lane (half, n=lane&15),
// element j -> k = kc*32 + 16*(j>>3) + 8*half + (j&7)
//   => two contiguous 8-half runs => 2x ds_load_b128 per fragment.
// Store epilogue: wave-uniform bounds split -> plain b32 stores in fast path.
// ---------------------------------------------------------------------------
#define WT_STRIDE 136   // halfs per LDS row (128 + 8 pad)

template <int RELU>
__global__ __launch_bounds__(256) void gat_gemm_wmma(
    const float* __restrict__ in, const float* __restrict__ Wg,
    float* __restrict__ out, int N)
{
    __shared__ __bf16 sWT[D * WT_STRIDE];   // ~34 KB of 320 KB WGP LDS

    // ---- pack W^T into LDS: thread t -> column n = t&127, k-slab (t>>7)*64.
    // Global reads W[k][n] are lane-coalesced along n; stores are 16B b128.
    {
        const int n     = threadIdx.x & 127;
        const int kbase = (threadIdx.x >> 7) * 64;
#pragma unroll
        for (int g = 0; g < 8; ++g) {
            const int k0 = kbase + g * 8;
            v8bf p;
#pragma unroll
            for (int j = 0; j < 8; ++j)
                p[j] = (__bf16)Wg[(k0 + j) * D + n];
            *(v8bf*)(&sWT[n * WT_STRIDE + k0]) = p;
        }
    }
    __syncthreads();

    const int wave    = threadIdx.x >> 5;
    const int lane    = threadIdx.x & 31;
    const int half    = lane >> 4;
    const int mrow    = lane & 15;
    const int rowBase = blockIdx.x * 128 + wave * 16;
    const bool fullTile = (rowBase + 16 <= N);   // wave-uniform

    // ---- A fragments: 16 rows x 128 K, four 16x32 chunks, b128 loads ----
    const int  arow   = rowBase + mrow;
    const bool avalid = arow < N;
    const float* xrow = in + (size_t)arow * D;
    const float4 f4z  = make_float4(0.f, 0.f, 0.f, 0.f);

    v16bf a[4];
#pragma unroll
    for (int kc = 0; kc < 4; ++kc) {
        const int kb = kc * 32 + half * 8;        // first 8-run
        float4 u0 = avalid ? *(const float4*)(xrow + kb)      : f4z;
        float4 u1 = avalid ? *(const float4*)(xrow + kb + 4)  : f4z;
        float4 u2 = avalid ? *(const float4*)(xrow + kb + 16) : f4z;  // second 8-run
        float4 u3 = avalid ? *(const float4*)(xrow + kb + 20) : f4z;
        float f[16] = { u0.x, u0.y, u0.z, u0.w, u1.x, u1.y, u1.z, u1.w,
                        u2.x, u2.y, u2.z, u2.w, u3.x, u3.y, u3.z, u3.w };
#pragma unroll
        for (int j = 0; j < 16; ++j) {
            float v = RELU ? fmaxf(f[j], 0.0f) : f[j];
            a[kc][j] = (__bf16)v;
        }
    }

    // ---- 8 output tiles of 16x16 along N ----
    const __bf16* wrow = &sWT[mrow * WT_STRIDE + half * 8];
#pragma unroll
    for (int nt = 0; nt < 8; ++nt) {
        const __bf16* wr = wrow + nt * 16 * WT_STRIDE;
        v8f c = {0.f, 0.f, 0.f, 0.f, 0.f, 0.f, 0.f, 0.f};
#pragma unroll
        for (int kc = 0; kc < 4; ++kc) {
            v8bf lo = *(const v8bf*)(wr + kc * 32);        // k run 0..7
            v8bf hi = *(const v8bf*)(wr + kc * 32 + 16);   // k run 16..23
            v16bf b = __builtin_shufflevector(lo, hi,
                        0, 1, 2, 3, 4, 5, 6, 7, 8, 9, 10, 11, 12, 13, 14, 15);
            c = __builtin_amdgcn_wmma_f32_16x16x32_bf16(
                    false, a[kc], false, b, (short)0, c, false, false);
        }
        // C/D layout: VGPR r -> row (half*8 + r), col = lane&15
        float* orow = out + (size_t)(rowBase + half * 8) * D + nt * 16 + mrow;
        if (fullTile) {
#pragma unroll
            for (int r = 0; r < 8; ++r)
                orow[(size_t)r * D] = c[r];
        } else {
#pragma unroll
            for (int r = 0; r < 8; ++r)
                if (rowBase + half * 8 + r < N) orow[(size_t)r * D] = c[r];
        }
    }
}

// ---------------------------------------------------------------------------
// Per-node logits: as[i] = h[i].a_src, ad[i] = h[i].a_dst.  Wave per row.
// ---------------------------------------------------------------------------
__global__ __launch_bounds__(256) void gat_logits(
    const float* __restrict__ h, const float* __restrict__ a_src,
    const float* __restrict__ a_dst, float* __restrict__ as_,
    float* __restrict__ ad_, int N)
{
    int wid  = (blockIdx.x * blockDim.x + threadIdx.x) >> 5;
    int lane = threadIdx.x & 31;
    if (wid >= N) return;

    float4 hv = *(const float4*)(h + (size_t)wid * D + lane * 4);
    float4 sv = *(const float4*)(a_src + lane * 4);
    float4 dv = *(const float4*)(a_dst + lane * 4);
    float ps = hv.x * sv.x + hv.y * sv.y + hv.z * sv.z + hv.w * sv.w;
    float pd = hv.x * dv.x + hv.y * dv.y + hv.z * dv.z + hv.w * dv.w;
#pragma unroll
    for (int off = 16; off > 0; off >>= 1) {
        ps += __shfl_down(ps, off, 32);
        pd += __shfl_down(pd, off, 32);
    }
    if (lane == 0) { as_[wid] = ps; ad_[wid] = pd; }
}

// ---------------------------------------------------------------------------
// Init: agg = bias broadcast, m = -inf key, z = 0
// ---------------------------------------------------------------------------
__global__ void gat_init(float* __restrict__ agg, const float* __restrict__ bias,
                         unsigned* __restrict__ m_u, float* __restrict__ z, int N)
{
    int tid = blockIdx.x * blockDim.x + threadIdx.x;
    if (tid < N * D) agg[tid] = bias[tid & (D - 1)];
    if (tid < N) { m_u[tid] = 0u; z[tid] = 0.0f; }
}

// ---------------------------------------------------------------------------
// Edge pass 1: e = leaky_relu(as[src] + ad[dst]); segment max via atomicMax
// ---------------------------------------------------------------------------
__global__ void gat_edge_max(
    const int* __restrict__ ei, const float* __restrict__ as_,
    const float* __restrict__ ad_, float* __restrict__ ee,
    unsigned* __restrict__ m_u, int E0, int N)
{
    int e = blockIdx.x * blockDim.x + threadIdx.x;
    int Etot = E0 + N;
    if (e >= Etot) return;
    int s, t;
    if (e < E0) { s = ei[e]; t = ei[E0 + e]; }
    else        { s = e - E0; t = s; }          // self loop
    float v = as_[s] + ad_[t];
    v = v > 0.0f ? v : NEG_SLOPE * v;
    ee[e] = v;
    atomicMax(&m_u[t], f2key(v));
}

// ---------------------------------------------------------------------------
// Edge pass 2: w = exp(e - m[dst]); z[dst] += w  (w overwrites ee)
// ---------------------------------------------------------------------------
__global__ void gat_edge_expsum(
    const int* __restrict__ ei, float* __restrict__ ee_in_w_out,
    const unsigned* __restrict__ m_u, float* __restrict__ z, int E0, int N)
{
    int e = blockIdx.x * blockDim.x + threadIdx.x;
    int Etot = E0 + N;
    if (e >= Etot) return;
    int t = (e < E0) ? ei[E0 + e] : (e - E0);
    float w = __expf(ee_in_w_out[e] - key2f(m_u[t]));
    ee_in_w_out[e] = w;
    atomicAddF32(&z[t], w);
}

// ---------------------------------------------------------------------------
// Edge pass 3: agg[dst] += (w/z[dst]) * h[src].  Wave per edge, float4/lane.
// ---------------------------------------------------------------------------
__global__ __launch_bounds__(256) void gat_edge_aggregate(
    const int* __restrict__ ei, const float* __restrict__ h,
    const float* __restrict__ w, const float* __restrict__ z,
    float* __restrict__ agg, int E0, int N)
{
    int gw   = (blockIdx.x * blockDim.x + threadIdx.x) >> 5;
    int lane = threadIdx.x & 31;
    int Etot = E0 + N;
    if (gw >= Etot) return;
    int s, t;
    if (gw < E0) { s = ei[gw]; t = ei[E0 + gw]; }
    else         { s = gw - E0; t = s; }
    float alpha = w[gw] / z[t];
    float4 hv = *(const float4*)(h + (size_t)s * D + lane * 4);
    float* dst = agg + (size_t)t * D + lane * 4;
    atomicAddF32(dst + 0, alpha * hv.x);
    atomicAddF32(dst + 1, alpha * hv.y);
    atomicAddF32(dst + 2, alpha * hv.z);
    atomicAddF32(dst + 3, alpha * hv.w);
}

__global__ void gat_relu(float* __restrict__ x, int total) {
    int tid = blockIdx.x * blockDim.x + threadIdx.x;
    if (tid < total) x[tid] = fmaxf(x[tid], 0.0f);
}

// ---------------------------------------------------------------------------
// Host launcher
// ---------------------------------------------------------------------------
extern "C" void kernel_launch(void* const* d_in, const int* in_sizes, int n_in,
                              void* d_out, int out_size, void* d_ws, size_t ws_size,
                              hipStream_t stream)
{
    const float* x      = (const float*)d_in[0];
    const int*   ei     = (const int*)d_in[1];
    const float* W1     = (const float*)d_in[2];
    const float* a_src1 = (const float*)d_in[3];
    const float* a_dst1 = (const float*)d_in[4];
    const float* b1     = (const float*)d_in[5];
    const float* W2     = (const float*)d_in[6];
    const float* a_src2 = (const float*)d_in[7];
    const float* a_dst2 = (const float*)d_in[8];
    const float* b2     = (const float*)d_in[9];

    const int N    = in_sizes[0] / D;     // 100000
    const int E0   = in_sizes[1] / 2;     // 625000
    const int Etot = E0 + N;

    // workspace carve-up (256B aligned)
    char* ws = (char*)d_ws;
    size_t off = 0;
    auto alloc = [&](size_t bytes) {
        char* p = ws + off;
        off = (off + bytes + 255) & ~(size_t)255;
        return p;
    };
    float*    h    = (float*)alloc((size_t)N * D * 4);  // reused by both layers
    float*    agg1 = (float*)alloc((size_t)N * D * 4);  // layer-1 output (pre-relu)
    float*    as_  = (float*)alloc((size_t)N * 4);
    float*    ad_  = (float*)alloc((size_t)N * 4);
    float*    z    = (float*)alloc((size_t)N * 4);
    unsigned* m_u  = (unsigned*)alloc((size_t)N * 4);
    float*    ee   = (float*)alloc((size_t)Etot * 4);

    const int gemmBlocks  = (N + 127) / 128;
    const int initBlocks  = (N * D + 255) / 256;
    const int rowWaveBlk  = (N * 32 + 255) / 256;             // wave per row
    const int edgeBlocks  = (Etot + 255) / 256;               // thread per edge
    const int edgeWaveBlk = (int)(((size_t)Etot * 32 + 255) / 256); // wave per edge

    float* out = (float*)d_out;

    // ---- layer 1: x -> agg1 (relu deferred to layer-2 input load) ----
    gat_gemm_wmma<0><<<gemmBlocks, 256, 0, stream>>>(x, W1, h, N);
    gat_init<<<initBlocks, 256, 0, stream>>>(agg1, b1, m_u, z, N);
    gat_logits<<<rowWaveBlk, 256, 0, stream>>>(h, a_src1, a_dst1, as_, ad_, N);
    gat_edge_max<<<edgeBlocks, 256, 0, stream>>>(ei, as_, ad_, ee, m_u, E0, N);
    gat_edge_expsum<<<edgeBlocks, 256, 0, stream>>>(ei, ee, m_u, z, E0, N);
    gat_edge_aggregate<<<edgeWaveBlk, 256, 0, stream>>>(ei, h, ee, z, agg1, E0, N);

    // ---- layer 2: relu(agg1) -> d_out ----
    gat_gemm_wmma<1><<<gemmBlocks, 256, 0, stream>>>(agg1, W2, h, N);
    gat_init<<<initBlocks, 256, 0, stream>>>(out, b2, m_u, z, N);
    gat_logits<<<rowWaveBlk, 256, 0, stream>>>(h, a_src2, a_dst2, as_, ad_, N);
    gat_edge_max<<<edgeBlocks, 256, 0, stream>>>(ei, as_, ad_, ee, m_u, E0, N);
    gat_edge_expsum<<<edgeBlocks, 256, 0, stream>>>(ei, ee, m_u, z, E0, N);
    gat_edge_aggregate<<<edgeWaveBlk, 256, 0, stream>>>(ei, h, ee, z, out, E0, N);

    gat_relu<<<initBlocks, 256, 0, stream>>>(out, N * D);
}
